// PAM_Module_68839735820808
// MI455X (gfx1250) — compile-verified, matrix operand order checked by reading, not run on hardware
//
#include <hip/hip_runtime.h>

#define B_  4
#define C_  512
#define CQ_ 64
#define N_  4096   // 64*64

typedef _Float16 v16h __attribute__((ext_vector_type(16)));
typedef _Float16 h8   __attribute__((ext_vector_type(8)));
typedef float    v8f  __attribute__((ext_vector_type(8)));
typedef float    f4   __attribute__((ext_vector_type(4)));

// Per the CDNA5 ISA 16-bit A/B VGPR layout (16x32, wave32):
//   lanes 0-15 : K = {0..7, 16..23}, lanes 16-31 : K = {8..15, 24..31}
// element e of the v16h: K(e) = kbase + e + (e>=8 ? 8 : 0), kbase = (lane>=16)?8:0
__device__ __forceinline__ int kmap(int e, int lane) {
    int kbase = (lane & 16) ? 8 : 0;
    return kbase + e + ((e < 8) ? 0 : 8);
}

// ---------------------------------------------------------------------------
// Projection GEMM:  out = W(ODIM x C) * x(C x N) + bias, emitted as f16.
// One wave per 16x16 output tile, K-loop of 16 steps of wmma_f32_16x16x32_f16.
// TRANS==1 stores out transposed as [N][ODIM] (used for q and k); in that
// layout each lane's 8 results are contiguous -> single b128 store.
// ---------------------------------------------------------------------------
template<int ODIM, int TRANS>
__global__ void __launch_bounds__(32) proj_kernel(
    const float* __restrict__ W, const float* __restrict__ bias,
    const float* __restrict__ x, _Float16* __restrict__ out)
{
    const int lane  = threadIdx.x & 31;
    const int ntile = blockIdx.x;
    const int otile = blockIdx.y;
    const int b     = blockIdx.z;

    const int col   = lane & 15;            // output column (n) / A row (m)
    const int kbase = (lane & 16) ? 8 : 0;
    const int mhalf = (lane >> 4) * 8;      // C/D row offset per lane half

    const float* xb = x + (size_t)b * C_ * N_;
    const int n_glob = ntile * 16 + col;

    v8f acc = {};
    for (int kk = 0; kk < C_ / 32; ++kk) {
        const int k0 = kk * 32;

        // A: weight tile, row (otile*16+col), contiguous K runs -> float4 loads
        const float* wr = W + (size_t)(otile * 16 + col) * C_ + k0 + kbase;
        f4 w0 = *(const f4*)(wr);
        f4 w1 = *(const f4*)(wr + 4);
        f4 w2 = *(const f4*)(wr + 16);
        f4 w3 = *(const f4*)(wr + 20);
        v16h a;
#pragma unroll
        for (int i = 0; i < 4; ++i) {
            a[i]      = (_Float16)w0[i];
            a[4 + i]  = (_Float16)w1[i];
            a[8 + i]  = (_Float16)w2[i];
            a[12 + i] = (_Float16)w3[i];
        }

        // B: x tile, column n fixed, K=c strided by N (coalesced across lanes)
        v16h bt;
#pragma unroll
        for (int e = 0; e < 16; ++e) {
            int c = k0 + kmap(e, lane);
            bt[e] = (_Float16)xb[(size_t)c * N_ + n_glob];
        }

        acc = __builtin_amdgcn_wmma_f32_16x16x32_f16(
            false, a, false, bt, (short)0, acc, false, false);
    }

    if (TRANS) {
        // out[(b*N + n)*ODIM + otile*16 + mhalf + j], j contiguous -> b128
        h8 vals;
#pragma unroll
        for (int j = 0; j < 8; ++j)
            vals[j] = (_Float16)(acc[j] + bias[otile * 16 + mhalf + j]);
        *(h8*)(out + ((size_t)b * N_ + n_glob) * ODIM + otile * 16 + mhalf) = vals;
    } else {
#pragma unroll
        for (int j = 0; j < 8; ++j) {
            int mg = otile * 16 + mhalf + j;
            out[((size_t)b * ODIM + mg) * N_ + n_glob] =
                (_Float16)(acc[j] + bias[mg]);
        }
    }
}

// ---------------------------------------------------------------------------
// Fused attention: one block (8 waves) per (batch, 16-query tile).
// Phase 1 : S^T = K^T * Q  (M=m keys, N=n queries) -> raw logits into LDS f16,
//           per-n running max per wave (lane-pair reduce via shfl_xor 16).
// Phase 1b: block max, exp(S - gmax) in LDS (h8 vectorized), row sums via
//           ds_add_f32 atomics.
// Phase 2 : O[c,n] = sum_m v[c,m] * P[m,n]; 8 waves x 64 channels, P read
//           from LDS with b128 loads. Epilogue: gamma * O / rowsum + x.
// ---------------------------------------------------------------------------
__global__ void __launch_bounds__(256, 1) attn_kernel(
    const _Float16* __restrict__ qT,   // [B][N][CQ]
    const _Float16* __restrict__ kT,   // [B][N][CQ]
    const _Float16* __restrict__ vbuf, // [B][C][N]
    const float* __restrict__ x,
    const float* __restrict__ gamma,
    float* __restrict__ out)
{
    extern __shared__ char smem_raw[];
    _Float16* P    = (_Float16*)smem_raw;              // 16 x 4096 f16 = 128 KB
    float*    wmax = (float*)(smem_raw + 16 * N_ * 2); // [8][16]
    float*    gmax = wmax + 128;                       // [16]
    float*    gsum = gmax + 16;                        // [16]

    const int tid   = threadIdx.x;
    const int wave  = tid >> 5;
    const int lane  = tid & 31;
    const int ntile = blockIdx.x;
    const int b     = blockIdx.y;

    const int col   = lane & 15;
    const int kb16  = (lane & 16) ? 8 : 0;
    const int mhalf = (lane >> 4) * 8;
    const int n_glob = ntile * 16 + col;

    // Q tile (B-matrix for both K-steps), contiguous b128 loads
    const _Float16* qrow = qT + ((size_t)b * N_ + n_glob) * CQ_;
    v16h qb[2];
#pragma unroll
    for (int s = 0; s < 2; ++s) {
        h8 lo = *(const h8*)(qrow + s * 32 + kb16);
        h8 hi = *(const h8*)(qrow + s * 32 + kb16 + 16);
#pragma unroll
        for (int i = 0; i < 8; ++i) { qb[s][i] = lo[i]; qb[s][8 + i] = hi[i]; }
    }

    // ---- Phase 1: logits for this wave's 512 keys ----
    const _Float16* kTb = kT + (size_t)b * N_ * CQ_;
    float lmax = -3.0e38f;
    for (int t = 0; t < 32; ++t) {
        const int m0 = wave * 512 + t * 16;
        const _Float16* krow = kTb + (size_t)(m0 + col) * CQ_; // A row = key m
        v8f acc = {};
#pragma unroll
        for (int s = 0; s < 2; ++s) {
            h8 lo = *(const h8*)(krow + s * 32 + kb16);
            h8 hi = *(const h8*)(krow + s * 32 + kb16 + 16);
            v16h a;
#pragma unroll
            for (int i = 0; i < 8; ++i) { a[i] = lo[i]; a[8 + i] = hi[i]; }
            acc = __builtin_amdgcn_wmma_f32_16x16x32_f16(
                false, a, false, qb[s], (short)0, acc, false, false);
        }
        h8 sv16;
#pragma unroll
        for (int j = 0; j < 8; ++j) {
            float sv = acc[j];
            lmax = fmaxf(lmax, sv);
            sv16[j] = (_Float16)sv;
        }
        // 8 contiguous halves, 16B aligned -> ds_store_b128
        *(h8*)(P + col * N_ + m0 + mhalf) = sv16;
    }
    lmax = fmaxf(lmax, __shfl_xor(lmax, 16, 32)); // combine lane pair: full n column
    if (lane < 16) wmax[wave * 16 + lane] = lmax;
    __syncthreads();

    // ---- Phase 1b: block max, exponentiate in LDS, row sums ----
    if (tid < 16) {
        float g = wmax[tid];
#pragma unroll
        for (int w2 = 1; w2 < 8; ++w2) g = fmaxf(g, wmax[w2 * 16 + tid]);
        gmax[tid] = g;
        gsum[tid] = 0.0f;
    }
    __syncthreads();
    {
        const int row = tid >> 4;          // query row n-local (0..15)
        const int seg = tid & 15;          // 256-key segment
        const float gm = gmax[row];
        _Float16* pr = P + row * N_ + seg * 256;
        float ls = 0.0f;
        for (int i = 0; i < 32; ++i) {     // 32 chunks of 8 halves (b128)
            h8 v = *(const h8*)(pr + i * 8);
            h8 pv;
#pragma unroll
            for (int e = 0; e < 8; ++e) {
                float p = __expf((float)v[e] - gm);
                pv[e] = (_Float16)p;
                ls += p;
            }
            *(h8*)(pr + i * 8) = pv;
        }
        atomicAdd(&gsum[row], ls);
    }
    __syncthreads();

    // ---- Phase 2: O = V * P, this wave owns channels [wave*64, wave*64+64) ----
    const _Float16* vb = vbuf + (size_t)b * C_ * N_;
    v8f z = {};
    v8f acc2[4] = {z, z, z, z};
    for (int t2 = 0; t2 < 128; ++t2) {
        const int mb = t2 * 32;
        // B tile from LDS: column n, K=m (two b128 LDS loads)
        h8 plo = *(const h8*)(P + col * N_ + mb + kb16);
        h8 phi = *(const h8*)(P + col * N_ + mb + kb16 + 16);
        v16h bt;
#pragma unroll
        for (int i = 0; i < 8; ++i) { bt[i] = plo[i]; bt[8 + i] = phi[i]; }
#pragma unroll
        for (int ct = 0; ct < 4; ++ct) {
            const int c = wave * 64 + ct * 16 + col;      // A row = channel
            const _Float16* vr = vb + (size_t)c * N_ + mb + kb16;
            h8 alo = *(const h8*)(vr);
            h8 ahi = *(const h8*)(vr + 16);
            v16h a;
#pragma unroll
            for (int i = 0; i < 8; ++i) { a[i] = alo[i]; a[8 + i] = ahi[i]; }
            acc2[ct] = __builtin_amdgcn_wmma_f32_16x16x32_f16(
                false, a, false, bt, (short)0, acc2[ct], false, false);
        }
    }

    // ---- Epilogue: gamma * O / rowsum + residual ----
    const float inv = 1.0f / gsum[col];
    const float g   = gamma[0];
    const float* xb = x + (size_t)b * C_ * N_;
    float*      ob  = out + (size_t)b * C_ * N_;
#pragma unroll
    for (int ct = 0; ct < 4; ++ct) {
#pragma unroll
        for (int j = 0; j < 8; ++j) {
            int c = wave * 64 + ct * 16 + mhalf + j;
            size_t idx = (size_t)c * N_ + n_glob;
            ob[idx] = g * acc2[ct][j] * inv + xb[idx];
        }
    }
}

// ---------------------------------------------------------------------------
extern "C" void kernel_launch(void* const* d_in, const int* in_sizes, int n_in,
                              void* d_out, int out_size, void* d_ws, size_t ws_size,
                              hipStream_t stream) {
    const float* x     = (const float*)d_in[0];
    const float* wq    = (const float*)d_in[1];
    const float* bq    = (const float*)d_in[2];
    const float* wk    = (const float*)d_in[3];
    const float* bk    = (const float*)d_in[4];
    const float* wv    = (const float*)d_in[5];
    const float* bv    = (const float*)d_in[6];
    const float* gamma = (const float*)d_in[7];
    float* out = (float*)d_out;

    // Workspace: qT (2MB) | kT (2MB) | v (16MB), all f16
    _Float16* qT = (_Float16*)d_ws;
    _Float16* kT = qT + (size_t)B_ * N_ * CQ_;
    _Float16* vb = kT + (size_t)B_ * N_ * CQ_;

    proj_kernel<CQ_, 1><<<dim3(N_ / 16, CQ_ / 16, B_), dim3(32), 0, stream>>>(wq, bq, x, qT);
    proj_kernel<CQ_, 1><<<dim3(N_ / 16, CQ_ / 16, B_), dim3(32), 0, stream>>>(wk, bk, x, kT);
    proj_kernel<C_,  0><<<dim3(N_ / 16, C_  / 16, B_), dim3(32), 0, stream>>>(wv, bv, x, vb);

    size_t smem = (size_t)16 * N_ * 2 + (128 + 16 + 16) * sizeof(float);
    attn_kernel<<<dim3(N_ / 16, B_), dim3(256), smem, stream>>>(qT, kT, vb, x, gamma, out);
}
